// TemporalGraphTransformer_57423712747659
// MI455X (gfx1250) — matrix operand order repaired and checked by reading, not compiled, not run on hardware
//
#include <hip/hip_runtime.h>
#include <hip/hip_bf16.h>

// ---------------------------------------------------------------------------
// Temporal Graph Transformer for MI455X (gfx1250, wave32, WMMA)
// bf16 pre-converted operands -> packed b128 A-fragment loads -> unrolled
// v_wmma_f32_16x16x32_bf16 ; edge softmax via atomics.
// ---------------------------------------------------------------------------

#define TGT_N      20000
#define TGT_E      320000
#define TGT_F      256
#define TGT_H      64
#define TGT_HEADS  4
#define TGT_DH     64
#define TGT_L      3
#define TGT_G      64
#define TGT_C      10
#define TGT_HD     (TGT_HEADS * TGT_DH)   // 256

typedef unsigned short u16;
typedef __attribute__((ext_vector_type(16))) __bf16 v16bf;
typedef __attribute__((ext_vector_type(8)))  float  v8f;

union Frag { v16bf v; uint4 q[2]; u16 u[16]; };

__device__ __forceinline__ u16 tgt_f2bf(float f) {
  unsigned int x = __float_as_uint(f);
  unsigned int r = 0x7FFFu + ((x >> 16) & 1u);   // round to nearest even
  return (u16)((x + r) >> 16);
}

__device__ __forceinline__ void tgt_atomic_max_f32(float* addr, float val) {
  if (val >= 0.0f) {
    atomicMax((int*)addr, __float_as_int(val));
  } else {
    atomicMin((unsigned int*)addr, __float_as_uint(val));
  }
}

// ---------------------------------------------------------------------------
// C[M x Nout] = A[M x K](bf16) @ B[K x Nout](bf16) + bias (+ bias2), fp32 acc.
// One wave computes one 16x16*NT tile strip; A fragment reused across NT.
// A loads are contiguous 16B chunks, B loads are per-u16 strided; both match ISA 7.12.2 layouts:
//   A 16x32: lane L -> row L&15; chunk0 = K[grp*8 .. +7], chunk1 = K[16+grp*8 ..]
//   B 32x16: lane L -> col L&15; K[grp*8 .. +7], K[16+grp*8 ..] (per-u16 strided loads)
//   C/D    : lane L -> col L&15; VGPR v -> row grp*8 + v
// K % 32 == 0 (templated, fully unrolled); M % 16 == 0; Nout % 16 == 0.
// ---------------------------------------------------------------------------
template <int K, int NT>
__global__ void tgt_gemm_bf16_kernel(const u16* __restrict__ A,
                                     const u16* __restrict__ B,
                                     int Nout,
                                     const float* __restrict__ bias,
                                     const float* __restrict__ bias2,
                                     float* __restrict__ C,
                                     u16* __restrict__ Cbf) {
  const int lane   = threadIdx.x;                         // 0..31
  const int tileM  = blockIdx.x;
  const int tileN0 = (blockIdx.y * 4 + threadIdx.y) * NT; // NT consecutive tiles
  if (tileN0 * 16 >= Nout) return;                        // uniform per wave

  const int r16 = lane & 15;
  const int grp = lane >> 4;
  const int row = tileM * 16 + r16;

  v8f acc[NT];
  const v8f vzero = {};
#pragma unroll
  for (int t = 0; t < NT; ++t) acc[t] = vzero;

#pragma unroll
  for (int kk = 0; kk < K; kk += 32) {
    Frag a;
    const u16* ap = A + (size_t)row * K + kk + grp * 8;
    a.q[0] = *(const uint4*)(ap);
    a.q[1] = *(const uint4*)(ap + 16);

    const int krow = kk + grp * 8;
    const u16* bp = B + (size_t)krow * Nout + tileN0 * 16 + r16;
#pragma unroll
    for (int t = 0; t < NT; ++t) {
      Frag b;
#pragma unroll
      for (int s = 0; s < 8; ++s) {
        b.u[s]     = bp[(size_t)s * Nout + t * 16];
        b.u[8 + s] = bp[(size_t)(16 + s) * Nout + t * 16];
      }
      acc[t] = __builtin_amdgcn_wmma_f32_16x16x32_bf16(
          /*neg_a=*/false, a.v, /*neg_b=*/false, b.v,
          /*c_mod=*/(short)0, acc[t], /*reuse_a=*/false, /*reuse_b=*/false);
    }
  }

#pragma unroll
  for (int t = 0; t < NT; ++t) {
    const int n = (tileN0 + t) * 16 + r16;
    float badd = (bias ? bias[n] : 0.0f) + (bias2 ? bias2[n] : 0.0f);
#pragma unroll
    for (int v = 0; v < 8; ++v) {
      const int m = tileM * 16 + grp * 8 + v;
      const float val = acc[t][v] + badd;
      C[(size_t)m * Nout + n] = val;
      if (Cbf) Cbf[(size_t)m * Nout + n] = tgt_f2bf(val);
    }
  }
}

// ---------------------------------------------------------------------------
__global__ void tgt_fill_kernel(float* __restrict__ p, long long n, float v) {
  long long i = (long long)blockIdx.x * blockDim.x + threadIdx.x;
  if (i < n) p[i] = v;
}

__global__ void tgt_cvt_bf16_kernel(const float* __restrict__ src,
                                    u16* __restrict__ dst, long long n) {
  long long i = (long long)blockIdx.x * blockDim.x + threadIdx.x;
  if (i < n) dst[i] = tgt_f2bf(src[i]);
}

// logits[e*4+h] = dot(q[dst,h,:], k[src,h,:]) / 8 ; atomic-max into m[dst*4+h]
__global__ void tgt_edge_logits_kernel(const int* __restrict__ ei,
                                       const float* __restrict__ q,
                                       const float* __restrict__ k,
                                       float* __restrict__ logits,
                                       float* __restrict__ m) {
  int t = blockIdx.x * blockDim.x + threadIdx.x;
  if (t >= TGT_E * TGT_HEADS) return;
  const int e = t >> 2, h = t & 3;
  const int src = ei[e], dst = ei[TGT_E + e];
  const float4* qp = (const float4*)(q + (size_t)dst * TGT_HD + h * TGT_DH);
  const float4* kp = (const float4*)(k + (size_t)src * TGT_HD + h * TGT_DH);
  float s = 0.0f;
#pragma unroll
  for (int i = 0; i < 16; ++i) {
    float4 a = qp[i], b = kp[i];
    s += a.x * b.x + a.y * b.y + a.z * b.z + a.w * b.w;
  }
  s *= 0.125f;  // 1/sqrt(DH=64)
  logits[t] = s;
  tgt_atomic_max_f32(&m[dst * TGT_HEADS + h], s);
}

// logits -> exp(logit - m[dst]) in place ; atomicAdd denom[dst]
__global__ void tgt_edge_exp_kernel(const int* __restrict__ ei,
                                    const float* __restrict__ m,
                                    float* __restrict__ logits,
                                    float* __restrict__ denom) {
  int t = blockIdx.x * blockDim.x + threadIdx.x;
  if (t >= TGT_E * TGT_HEADS) return;
  const int e = t >> 2, h = t & 3;
  const int dst = ei[TGT_E + e];
  float ev = __expf(logits[t] - m[dst * TGT_HEADS + h]);
  logits[t] = ev;
  atomicAdd(&denom[dst * TGT_HEADS + h], ev);
}

// agg[dst, c] += (e/denom) * v[src, c]   (c = head*64 + d, 256 per edge)
__global__ void tgt_edge_scatter_kernel(const int* __restrict__ ei,
                                        const float* __restrict__ evals,
                                        const float* __restrict__ denom,
                                        const float* __restrict__ vmat,
                                        float* __restrict__ agg) {
  long long t = (long long)blockIdx.x * blockDim.x + threadIdx.x;
  if (t >= (long long)TGT_E * TGT_HD) return;
  const int e = (int)(t >> 8);
  const int c = (int)(t & 255);
  const int h = c >> 6;
  const int src = ei[e], dst = ei[TGT_E + e];
  const float alpha = evals[e * TGT_HEADS + h] / denom[dst * TGT_HEADS + h];
  atomicAdd(&agg[(size_t)dst * TGT_HD + c], alpha * vmat[(size_t)src * TGT_HD + c]);
}

// h[n,c] = relu( mean_heads(agg[n,:,c]) + skip[n,c] ), also emit bf16 copy
__global__ void tgt_node_finalize_kernel(const float* __restrict__ agg,
                                         const float* __restrict__ skip,
                                         float* __restrict__ h,
                                         u16* __restrict__ hbf) {
  int t = blockIdx.x * blockDim.x + threadIdx.x;
  if (t >= TGT_N * TGT_H) return;
  const int n = t >> 6, c = t & 63;
  const float* ap = agg + (size_t)n * TGT_HD;
  float s = ap[c] + ap[64 + c] + ap[128 + c] + ap[192 + c];
  float val = 0.25f * s + skip[t];
  val = val > 0.0f ? val : 0.0f;
  h[t] = val;
  hbf[t] = tgt_f2bf(val);
}

// segment-sum pool into pool[g][branch][c] + counts
__global__ void tgt_pool_kernel(const float* __restrict__ h,
                                const int* __restrict__ batch, int t_idx,
                                float* __restrict__ pool, float* __restrict__ cnt) {
  int t = blockIdx.x * blockDim.x + threadIdx.x;
  if (t >= TGT_N * TGT_H) return;
  const int n = t >> 6, c = t & 63;
  const int g = batch[n];
  atomicAdd(&pool[((size_t)g * 2 + t_idx) * TGT_H + c], h[t]);
  if (c == 0) atomicAdd(&cnt[g * 2 + t_idx], 1.0f);
}

// whole head: mean-pool normalize -> MHA(seq=2) -> mean -> MLP. 1 block/graph.
__global__ void tgt_head_kernel(const float* __restrict__ pool,
                                const float* __restrict__ cnt,
                                const float* __restrict__ in_w,
                                const float* __restrict__ in_b,
                                const float* __restrict__ out_w,
                                const float* __restrict__ out_b,
                                const float* __restrict__ Wc1,
                                const float* __restrict__ bc1,
                                const float* __restrict__ Wc2,
                                const float* __restrict__ bc2,
                                float* __restrict__ out) {
  const int g = blockIdx.x;
  const int j = threadIdx.x;           // 0..63
  __shared__ float xg[2][64];
  __shared__ float qkv[2][192];
  __shared__ float o[2][64];
  __shared__ float xf[64];
  __shared__ float z1[32];
  __shared__ float attn[TGT_HEADS][2][2];

  for (int s = 0; s < 2; ++s) {
    float c = cnt[g * 2 + s];
    c = c > 1.0f ? c : 1.0f;
    xg[s][j] = pool[((size_t)g * 2 + s) * TGT_H + j] / c;
  }
  __syncthreads();

  for (int s = 0; s < 2; ++s)
    for (int jj = j; jj < 192; jj += 64) {
      float acc = in_b[jj];
      for (int c = 0; c < 64; ++c) acc += xg[s][c] * in_w[jj * 64 + c];
      qkv[s][jj] = acc;
    }
  __syncthreads();

  if (j < TGT_HEADS) {   // per-head 2x2 attention, dh = 16, scale 0.25
    float sc[2][2];
    for (int qs = 0; qs < 2; ++qs)
      for (int ks = 0; ks < 2; ++ks) {
        float a = 0.0f;
        for (int d = 0; d < 16; ++d)
          a += qkv[qs][j * 16 + d] * qkv[ks][64 + j * 16 + d];
        sc[qs][ks] = a * 0.25f;
      }
    for (int qs = 0; qs < 2; ++qs) {
      float mx = sc[qs][0] > sc[qs][1] ? sc[qs][0] : sc[qs][1];
      float e0 = __expf(sc[qs][0] - mx), e1 = __expf(sc[qs][1] - mx);
      float d = e0 + e1;
      attn[j][qs][0] = e0 / d;
      attn[j][qs][1] = e1 / d;
    }
  }
  __syncthreads();

  const int hh = j >> 4;
  for (int s = 0; s < 2; ++s)
    o[s][j] = attn[hh][s][0] * qkv[0][128 + j] + attn[hh][s][1] * qkv[1][128 + j];
  __syncthreads();

  {
    float y0 = out_b[j], y1 = out_b[j];
    for (int c = 0; c < 64; ++c) {
      float w = out_w[j * 64 + c];
      y0 += o[0][c] * w;
      y1 += o[1][c] * w;
    }
    xf[j] = 0.5f * (y0 + y1);
  }
  __syncthreads();

  if (j < 32) {
    float a = bc1[j];
    for (int c = 0; c < 64; ++c) a += xf[c] * Wc1[c * 32 + j];
    z1[j] = a > 0.0f ? a : 0.0f;
  }
  __syncthreads();

  if (j < TGT_C) {
    float a = bc2[j];
    for (int c = 0; c < 32; ++c) a += z1[c] * Wc2[c * TGT_C + j];
    out[g * TGT_C + j] = a;
  }
}

// ---------------------------------------------------------------------------
extern "C" void kernel_launch(void* const* d_in, const int* in_sizes, int n_in,
                              void* d_out, int out_size, void* d_ws, size_t ws_size,
                              hipStream_t stream) {
  (void)in_sizes; (void)n_in; (void)out_size; (void)ws_size;

  const float* x1    = (const float*)d_in[0];
  const float* x2    = (const float*)d_in[1];
  const int*   ei1   = (const int*)  d_in[2];
  const int*   ei2   = (const int*)  d_in[3];
  const int*   bat1  = (const int*)  d_in[4];
  const int*   bat2  = (const int*)  d_in[5];
  const float* Wemb  = (const float*)d_in[6];
  const float* bemb  = (const float*)d_in[7];
  const float* temb  = (const float*)d_in[8];
  const float* Wq    = (const float*)d_in[9];
  const float* bq    = (const float*)d_in[10];
  const float* Wk    = (const float*)d_in[11];
  const float* bk    = (const float*)d_in[12];
  const float* Wv    = (const float*)d_in[13];
  const float* bv    = (const float*)d_in[14];
  const float* Ws    = (const float*)d_in[15];
  const float* bs    = (const float*)d_in[16];
  const float* in_w  = (const float*)d_in[17];
  const float* in_b  = (const float*)d_in[18];
  const float* out_w = (const float*)d_in[19];
  const float* out_b = (const float*)d_in[20];
  const float* Wc1   = (const float*)d_in[21];
  const float* bc1   = (const float*)d_in[22];
  const float* Wc2   = (const float*)d_in[23];
  const float* bc2   = (const float*)d_in[24];
  float* out = (float*)d_out;

  // workspace layout (float-granular allocator, 16B-aligned chunks)
  float* ws = (float*)d_ws;
  size_t off = 0;
  auto alloc = [&](size_t n) { float* p = ws + off; off += (n + 3) & ~(size_t)3; return p; };
  float* h1     = alloc((size_t)TGT_N * TGT_H);
  float* h2     = alloc((size_t)TGT_N * TGT_H);
  float* qb     = alloc((size_t)TGT_N * TGT_HD);
  float* kb     = alloc((size_t)TGT_N * TGT_HD);
  float* vb     = alloc((size_t)TGT_N * TGT_HD);
  float* skip   = alloc((size_t)TGT_N * TGT_H);
  float* logits = alloc((size_t)TGT_E * TGT_HEADS);
  float* mmax   = alloc((size_t)TGT_N * TGT_HEADS);
  float* denom  = alloc((size_t)TGT_N * TGT_HEADS);
  float* agg    = alloc((size_t)TGT_N * TGT_HD);
  float* pool   = alloc((size_t)TGT_G * 2 * TGT_H);
  float* cnt    = alloc((size_t)TGT_G * 2);
  // bf16 buffers (2 u16 per float slot)
  u16* xbf    = (u16*)alloc(((size_t)TGT_N * TGT_F) / 2);
  u16* hbf    = (u16*)alloc(((size_t)TGT_N * TGT_H) / 2);
  u16* WembBf = (u16*)alloc(((size_t)TGT_F * TGT_H) / 2);
  u16* WqBf   = (u16*)alloc(((size_t)TGT_L * TGT_H * TGT_HD) / 2);
  u16* WkBf   = (u16*)alloc(((size_t)TGT_L * TGT_H * TGT_HD) / 2);
  u16* WvBf   = (u16*)alloc(((size_t)TGT_L * TGT_H * TGT_HD) / 2);
  u16* WsBf   = (u16*)alloc(((size_t)TGT_L * TGT_H * TGT_H) / 2);

  auto fill = [&](float* p, long long n, float v) {
    tgt_fill_kernel<<<(unsigned)((n + 255) / 256), 256, 0, stream>>>(p, n, v);
  };
  auto cvt = [&](const float* s, u16* d, long long n) {
    tgt_cvt_bf16_kernel<<<(unsigned)((n + 255) / 256), 256, 0, stream>>>(s, d, n);
  };

  const long long NH = (long long)TGT_N * TGT_H;
  const long long EH = (long long)TGT_E * TGT_HEADS;
  const long long EC = (long long)TGT_E * TGT_HD;
  const int MT = TGT_N / 16;  // 1250 M-tiles

  // weights -> bf16 (deterministic, done every call; tiny)
  cvt(Wemb, WembBf, (long long)TGT_F * TGT_H);
  cvt(Wq,   WqBf,   (long long)TGT_L * TGT_H * TGT_HD);
  cvt(Wk,   WkBf,   (long long)TGT_L * TGT_H * TGT_HD);
  cvt(Wv,   WvBf,   (long long)TGT_L * TGT_H * TGT_HD);
  cvt(Ws,   WsBf,   (long long)TGT_L * TGT_H * TGT_H);

  fill(pool, (long long)TGT_G * 2 * TGT_H, 0.0f);
  fill(cnt,  (long long)TGT_G * 2, 0.0f);

  const dim3 gblk(32, 4);

  for (int t = 0; t < 2; ++t) {
    const float* x   = (t == 0) ? x1 : x2;
    const int*   ei  = (t == 0) ? ei1 : ei2;
    const int*   bat = (t == 0) ? bat1 : bat2;
    float*       h   = (t == 0) ? h1 : h2;

    // x -> bf16, then h = x @ W_emb + b_emb + temp_emb[t] (fp32 + bf16 out)
    cvt(x, xbf, (long long)TGT_N * TGT_F);
    tgt_gemm_bf16_kernel<TGT_F, 1><<<dim3(MT, 1), gblk, 0, stream>>>(
        xbf, WembBf, TGT_H, bemb, temb + t * TGT_H, h, hbf);

    for (int l = 0; l < TGT_L; ++l) {
      const size_t wo = (size_t)l * TGT_H * TGT_HD;
      // Q/K/V: Nout=256 -> 16 N-tiles, NT=2 -> 8 tiles/block, grid.y=2
      tgt_gemm_bf16_kernel<TGT_H, 2><<<dim3(MT, 2), gblk, 0, stream>>>(
          hbf, WqBf + wo, TGT_HD, bq + (size_t)l * TGT_HD, nullptr, qb, nullptr);
      tgt_gemm_bf16_kernel<TGT_H, 2><<<dim3(MT, 2), gblk, 0, stream>>>(
          hbf, WkBf + wo, TGT_HD, bk + (size_t)l * TGT_HD, nullptr, kb, nullptr);
      tgt_gemm_bf16_kernel<TGT_H, 2><<<dim3(MT, 2), gblk, 0, stream>>>(
          hbf, WvBf + wo, TGT_HD, bv + (size_t)l * TGT_HD, nullptr, vb, nullptr);
      // skip: Nout=64 -> 4 N-tiles, NT=1, grid.y=1
      tgt_gemm_bf16_kernel<TGT_H, 1><<<dim3(MT, 1), gblk, 0, stream>>>(
          hbf, WsBf + (size_t)l * TGT_H * TGT_H, TGT_H,
          bs + (size_t)l * TGT_H, nullptr, skip, nullptr);

      fill(mmax,  (long long)TGT_N * TGT_HEADS, -__builtin_inff());
      fill(denom, (long long)TGT_N * TGT_HEADS, 0.0f);
      fill(agg,   (long long)TGT_N * TGT_HD, 0.0f);

      tgt_edge_logits_kernel<<<(unsigned)((EH + 255) / 256), 256, 0, stream>>>(
          ei, qb, kb, logits, mmax);
      tgt_edge_exp_kernel<<<(unsigned)((EH + 255) / 256), 256, 0, stream>>>(
          ei, mmax, logits, denom);
      tgt_edge_scatter_kernel<<<(unsigned)((EC + 255) / 256), 256, 0, stream>>>(
          ei, logits, denom, vb, agg);
      tgt_node_finalize_kernel<<<(unsigned)((NH + 255) / 256), 256, 0, stream>>>(
          agg, skip, h, hbf);
    }

    tgt_pool_kernel<<<(unsigned)((NH + 255) / 256), 256, 0, stream>>>(
        h, bat, t, pool, cnt);
  }

  tgt_head_kernel<<<TGT_G, TGT_H, 0, stream>>>(
      pool, cnt, in_w, in_b, out_w, out_b, Wc1, bc1, Wc2, bc2, out);
}
